// UVit_33964601377065
// MI455X (gfx1250) — matrix operand order repaired and checked by reading, not compiled, not run on hardware
//
#include <hip/hip_runtime.h>
#include <cstddef>

// ---------------------------------------------------------------------------
// U-ViT forward for MI455X (gfx1250, wave32, WMMA + dual-TDM pipelined GEMM).
// Weights are pre-packed to bf16 column-major (Wt[N][K]) so BOTH GEMM operand
// tiles are verbatim 2-D bf16 copies staged by the Tensor Data Mover into
// double-buffered LDS; the k-loop contains no manual global loads at all.
// WMMA operands load as per-lane ds_load_b128 pairs.
// ---------------------------------------------------------------------------

#define DEV __device__ __forceinline__

typedef __attribute__((ext_vector_type(16))) __bf16 v16bf;
typedef __attribute__((ext_vector_type(8)))  __bf16 v8bf;
typedef __attribute__((ext_vector_type(8)))  float  v8f;

#if __has_builtin(__builtin_amdgcn_tensor_load_to_lds) && \
    __has_builtin(__builtin_amdgcn_s_wait_tensorcnt)
#define USE_TDM 1
#else
#define USE_TDM 0
#endif

typedef unsigned int u32x4 __attribute__((ext_vector_type(4)));
typedef int          i32x4 __attribute__((ext_vector_type(4)));
typedef int          i32x8 __attribute__((ext_vector_type(8)));

static constexpr int Bv   = 4;
static constexpr int Dm   = 768;
static constexpr int NH   = 12;
static constexpr int FF   = 3072;
static constexpr int Lm   = 256;
static constexpr int MT   = Bv * Lm;               // 1024 token rows
static constexpr int NCONV = Bv * 128 * 256 * 256; // 33,554,432

DEV v8f wmma_bf16(v16bf a, v16bf b, v8f c) {
  return __builtin_amdgcn_wmma_f32_16x16x32_bf16(false, a, false, b,
                                                 (short)0, c, false, false);
}

// A fragment (16x32) from a ROW-MAJOR bf16 tile with row stride lda.
DEV v16bf load_a16(const __bf16* base, int lda) {
  int lane = threadIdx.x & 31;
  int h = lane >> 4, m = lane & 15;
  const __bf16* rp = base + m * lda + 16 * h;
  struct Pair { v8bf lo, hi; } p;
  p.lo = *(const v8bf*)rp;
  p.hi = *(const v8bf*)(rp + 8);
  return __builtin_bit_cast(v16bf, p);
}

// B fragment (32x16) from an N-MAJOR tile: element (k,n) at base[n*ldb + k].
DEV v16bf load_b16(const __bf16* base, int ldb) {
  int lane = threadIdx.x & 31;
  int h = lane >> 4, n = lane & 15;
  struct Pair { v8bf lo, hi; };
  Pair p = *(const Pair*)(base + n * ldb + 16 * h);
  return __builtin_bit_cast(v16bf, p);
}

#if USE_TDM
// Issue a TDM 2-D tile load: tile0 x tile1 bf16 elems from a dim0 x dim1
// tensor (row stride dim0 elems) into row-major LDS at byte offset `lds`.
DEV void tdm_load_2d(unsigned lds, const void* gaddr, int dim0, int dim1,
                     int tile0, int tile1) {
  unsigned long long ga = (unsigned long long)(size_t)gaddr;
  u32x4 g0;
  g0[0] = 1u;                                   // count=1, user mode
  g0[1] = lds;                                  // lds_addr (bytes)
  g0[2] = (unsigned)ga;                         // global_addr[31:0]
  g0[3] = (unsigned)((ga >> 32) & 0x01FFFFFFu)  // global_addr[56:32]
          | 0x80000000u;                        // type=2 (image)
  i32x8 g1;
  g1[0] = (int)(1u << 16);                      // data_size=2B; wg mask=0
  g1[1] = (int)(((unsigned)dim0 & 0xFFFFu) << 16);         // tensor_dim0 lo
  g1[2] = (int)((((unsigned)dim0 >> 16) & 0xFFFFu) |
                (((unsigned)dim1 & 0xFFFFu) << 16));       // dim0 hi|dim1 lo
  g1[3] = (int)((((unsigned)dim1 >> 16) & 0xFFFFu) |
                (((unsigned)tile0) << 16));                // dim1 hi|tile0
  g1[4] = tile1;                                // tile_dim1 (tile_dim2=0)
  g1[5] = dim0;                                 // tensor_dim0_stride lo
  g1[6] = 0;
  g1[7] = 0;
  i32x4 z4 = {0, 0, 0, 0};
  i32x8 z8 = {0, 0, 0, 0, 0, 0, 0, 0};
  __builtin_amdgcn_tensor_load_to_lds(g0, g1, z4, z4, z8, 0);
}
#endif

DEV float mish_f(float v) {
  float sp = (v > 20.f) ? v : log1pf(__expf(v));
  return v * tanhf(sp);
}

// ---------------------------------------------------------------------------
// Weight pre-pack: Wt[n*K + k] = bf16(W[k*N + n]) via LDS-tiled transpose.
// grid (N/32, K/32), 256 threads.
// ---------------------------------------------------------------------------
__global__ __launch_bounds__(256) void transpose_pack_kernel(
    const float* __restrict__ W, __bf16* __restrict__ Wt, int K, int N) {
  __shared__ float tile[32][33];
  int n0 = blockIdx.x * 32, k0 = blockIdx.y * 32;
  int tx = threadIdx.x & 31, ty = threadIdx.x >> 5;  // ty: 0..7
#pragma unroll
  for (int j = 0; j < 4; ++j)
    tile[ty + 8 * j][tx] = W[(size_t)(k0 + ty + 8 * j) * N + n0 + tx];
  __syncthreads();
#pragma unroll
  for (int j = 0; j < 4; ++j)
    Wt[(size_t)(n0 + ty + 8 * j) * K + k0 + tx] = (__bf16)tile[tx][ty + 8 * j];
}

// ---------------------------------------------------------------------------
// GEMM: out[M,N](f32) = A_bf16[M,K] @ Wt_bf16[N,K]^T (+bias[N]) (+resid)
// Block tile 64x128, 8 waves -> each wave 16x64. M%64==0, N%128==0, K%32==0.
// Both LDS tiles staged by TDM (A: wave0, B: wave1), double buffered, one
// barrier per k-step; DMA for step i+1 overlaps WMMA on step i.
// ---------------------------------------------------------------------------
__global__ __launch_bounds__(256) void gemm_kernel(
    const __bf16* __restrict__ A, const __bf16* __restrict__ Wt,
    const float* __restrict__ bias, const float* __restrict__ resid,
    float* __restrict__ out, int M, int N, int K) {
  __shared__ __align__(32) __bf16 As[2][64 * 32];   // row-major A tiles
  __shared__ __align__(32) __bf16 Bt[2][128 * 32];  // n-major B tiles
  int m0 = blockIdx.y * 64, n0 = blockIdx.x * 128;
  int tid = threadIdx.x;
  int w = tid >> 5;
  int wrt = w & 3;         // wave row-tile (16 rows)
  int wns = (w >> 2) * 4;  // wave first n-subtile (of 8)
  v8f acc[4] = {{}, {}, {}, {}};
  int nsteps = K >> 5;

#if USE_TDM
  if (tid < 32)
    tdm_load_2d((unsigned)(size_t)&As[0][0], &A[(size_t)m0 * K], K, M, 32, 64);
  else if (tid < 64)
    tdm_load_2d((unsigned)(size_t)&Bt[0][0], &Wt[(size_t)n0 * K], K, N, 32, 128);

  for (int i = 0; i < nsteps; ++i) {
    int k0 = i << 5;
    if (tid < 64) __builtin_amdgcn_s_wait_tensorcnt(0);
    __syncthreads();  // tile i visible; compute i-1 fully drained
    if (i + 1 < nsteps) {
      int kn = k0 + 32;
      if (tid < 32)
        tdm_load_2d((unsigned)(size_t)&As[(i + 1) & 1][0],
                    &A[(size_t)m0 * K + kn], K, M, 32, 64);
      else if (tid < 64)
        tdm_load_2d((unsigned)(size_t)&Bt[(i + 1) & 1][0],
                    &Wt[(size_t)n0 * K + kn], K, N, 32, 128);
      if (k0 + 64 < K)  // warm L2 for the slab after next
        __builtin_prefetch(&Wt[(size_t)(n0 + (tid & 127)) * K + k0 + 64], 0, 1);
    }
    const __bf16* as = &As[i & 1][0];
    const __bf16* bt = &Bt[i & 1][0];
    v16bf a = load_a16(as + wrt * 16 * 32, 32);
#pragma unroll
    for (int t = 0; t < 4; ++t) {
      v16bf b = load_b16(bt + (wns + t) * 16 * 32, 32);
      acc[t] = wmma_bf16(a, b, acc[t]);
    }
  }
#else
  for (int i = 0; i < nsteps; ++i) {
    int k0 = i << 5;
    {  // A stage: contiguous 16B copies
      int m = tid >> 2, kg = tid & 3;
      *(v8bf*)&As[0][m * 32 + kg * 8] =
          *(const v8bf*)&A[(size_t)(m0 + m) * K + k0 + kg * 8];
    }
    for (int ci = tid; ci < 512; ci += 256) {  // B stage: contiguous copies
      int n = ci >> 2, kg = ci & 3;
      *(v8bf*)&Bt[0][n * 32 + kg * 8] =
          *(const v8bf*)&Wt[(size_t)(n0 + n) * K + k0 + kg * 8];
    }
    __syncthreads();
    v16bf a = load_a16(&As[0][wrt * 16 * 32], 32);
#pragma unroll
    for (int t = 0; t < 4; ++t) {
      v16bf b = load_b16(&Bt[0][(wns + t) * 16 * 32], 32);
      acc[t] = wmma_bf16(a, b, acc[t]);
    }
    __syncthreads();
  }
#endif

  int lane = tid & 31, h = lane >> 4, n16 = lane & 15;
#pragma unroll
  for (int t = 0; t < 4; ++t) {
#pragma unroll
    for (int r = 0; r < 8; ++r) {
      int row = m0 + wrt * 16 + r + 8 * h;
      int col = n0 + (wns + t) * 16 + n16;
      float v = acc[t][r];
      if (bias) v += bias[col];
      size_t idx = (size_t)row * N + col;
      if (resid) v += resid[idx];
      out[idx] = v;
    }
  }
}

// ---------------------------------------------------------------------------
// Attention: one block per (batch, head). qkv f32 [1024, 2304] row-major.
// Reference scales q and k each by 1/sqrt(HD) -> fold 1/64 into Q.
// ---------------------------------------------------------------------------
__global__ __launch_bounds__(256) void attn_kernel(const float* __restrict__ qkv,
                                                   __bf16* __restrict__ attnO) {
  extern __shared__ __align__(32) char smem[];
  __bf16* Qs = (__bf16*)smem;    // [256 q][64 d] row-major (scaled)
  __bf16* Ks = Qs + 256 * 64;    // [256 key][64 d] row-major
  __bf16* Vt = Ks + 256 * 64;    // [64 d][256 key] transposed
  __bf16* Pall = Vt + 64 * 256;  // per wave: [16 q][256 key] row-major

  int bh = blockIdx.x;  // 0..47
  int b = bh / NH, hh = bh % NH;
  int tid = threadIdx.x, w = tid >> 5, lane = tid & 31;
  __bf16* Pw = Pall + w * 16 * 256;
  const float* qbase = qkv + (size_t)b * 256 * 2304 + (size_t)hh * 64;

  for (int ci = tid; ci < 2048; ci += 256) {
    int q = ci >> 3, dg = ci & 7;
    const float* src = qbase + (size_t)q * 2304 + dg * 8;
    v8bf vq, vk;
#pragma unroll
    for (int j = 0; j < 8; ++j) vq[j] = (__bf16)(src[j] * (1.0f / 64.0f));
#pragma unroll
    for (int j = 0; j < 8; ++j) vk[j] = (__bf16)src[768 + j];
    *(v8bf*)&Qs[q * 64 + dg * 8] = vq;
    *(v8bf*)&Ks[q * 64 + dg * 8] = vk;
  }
  for (int ci = tid; ci < 1024; ci += 256) {
    int n64 = ci & 63, kg = ci >> 6;
    const float* src = qbase + (size_t)(kg * 16) * 2304 + 1536 + n64;
    v8bf lo, hi;
#pragma unroll
    for (int k = 0; k < 8; ++k) lo[k] = (__bf16)src[(size_t)k * 2304];
#pragma unroll
    for (int k = 0; k < 8; ++k) hi[k] = (__bf16)src[(size_t)(k + 8) * 2304];
    __bf16* dst = &Vt[n64 * 256 + kg * 16];
    *(v8bf*)dst = lo;
    *(v8bf*)(dst + 8) = hi;
  }
  __syncthreads();

  int h16 = lane >> 4, n16 = lane & 15;
  for (int qi = 0; qi < 2; ++qi) {
    int qt = 2 * w + qi;
    int qb = qt * 16;
    v16bf aq0 = load_a16(&Qs[qb * 64], 64);
    v16bf aq1 = load_a16(&Qs[qb * 64 + 32], 64);

    v8f acc[16] = {};
#pragma unroll
    for (int nt = 0; nt < 16; ++nt) {
      acc[nt] = wmma_bf16(aq0, load_b16(&Ks[nt * 16 * 64], 64), acc[nt]);
      acc[nt] = wmma_bf16(aq1, load_b16(&Ks[nt * 16 * 64 + 32], 64), acc[nt]);
    }

    float inv[8];
#pragma unroll
    for (int r = 0; r < 8; ++r) {
      float m = acc[0][r];
#pragma unroll
      for (int nt = 1; nt < 16; ++nt) m = fmaxf(m, acc[nt][r]);
#pragma unroll
      for (int mask = 8; mask > 0; mask >>= 1) m = fmaxf(m, __shfl_xor(m, mask, 32));
      float s = 0.f;
#pragma unroll
      for (int nt = 0; nt < 16; ++nt) {
        float e = __expf(acc[nt][r] - m);
        acc[nt][r] = e;
        s += e;
      }
#pragma unroll
      for (int mask = 8; mask > 0; mask >>= 1) s += __shfl_xor(s, mask, 32);
      inv[r] = 1.0f / s;
    }

#pragma unroll
    for (int nt = 0; nt < 16; ++nt)
#pragma unroll
      for (int r = 0; r < 8; ++r)
        Pw[(r + 8 * h16) * 256 + nt * 16 + n16] = (__bf16)(acc[nt][r] * inv[r]);
    __syncthreads();

    v16bf ap[8];
#pragma unroll
    for (int kk = 0; kk < 8; ++kk) ap[kk] = load_a16(&Pw[kk * 32], 256);
#pragma unroll
    for (int dt = 0; dt < 4; ++dt) {
      v8f o = {};
#pragma unroll
      for (int kk = 0; kk < 8; ++kk)
        o = wmma_bf16(ap[kk], load_b16(&Vt[dt * 16 * 256 + kk * 32], 256), o);
#pragma unroll
      for (int r = 0; r < 8; ++r) {
        attnO[(size_t)(b * 256 + qb + r + 8 * h16) * 768 + hh * 64 + dt * 16 + n16] =
            (__bf16)o[r];
      }
    }
    __syncthreads();
  }
}

// ---------------------------------------------------------------------------
// Patchify + RMS norm -> bf16 A buffer. One block per token row (1024 rows).
// ---------------------------------------------------------------------------
__global__ __launch_bounds__(256) void patchify_rms_kernel(
    const float* __restrict__ x, const float* __restrict__ w,
    __bf16* __restrict__ out) {
  __shared__ float red[256];
  int row = blockIdx.x;
  int b = row >> 8, l = row & 255;
  int gy = l >> 4, gx = l & 15;
  int tid = threadIdx.x;
  float vals[3];
  float ss = 0.f;
#pragma unroll
  for (int i = 0; i < 3; ++i) {
    int pd = tid * 3 + i;
    int c = pd % 3, pp = pd / 3;
    int p1 = pp >> 4, p2 = pp & 15;
    float v = x[((size_t)(b * 3 + c) * 256 + gy * 16 + p1) * 256 + gx * 16 + p2];
    vals[i] = v;
    ss += v * v;
  }
  red[tid] = ss;
  __syncthreads();
  for (int s = 128; s > 0; s >>= 1) {
    if (tid < s) red[tid] += red[tid + s];
    __syncthreads();
  }
  float scale = rsqrtf(red[0] / 768.0f + 1e-5f);
#pragma unroll
  for (int i = 0; i < 3; ++i) {
    int pd = tid * 3 + i;
    out[(size_t)row * 768 + pd] = (__bf16)(vals[i] * scale * w[pd]);
  }
}

// RMS norm over 768 cols: f32 in -> bf16 out. One block per row.
__global__ __launch_bounds__(256) void rms_kernel(const float* __restrict__ in,
                                                  const float* __restrict__ w,
                                                  __bf16* __restrict__ out) {
  __shared__ float red[256];
  int row = blockIdx.x, tid = threadIdx.x;
  const float* rp = in + (size_t)row * 768;
  float vals[3];
  float ss = 0.f;
#pragma unroll
  for (int i = 0; i < 3; ++i) {
    float v = rp[tid * 3 + i];
    vals[i] = v;
    ss += v * v;
  }
  red[tid] = ss;
  __syncthreads();
  for (int s = 128; s > 0; s >>= 1) {
    if (tid < s) red[tid] += red[tid + s];
    __syncthreads();
  }
  float scale = rsqrtf(red[0] / 768.0f + 1e-5f);
#pragma unroll
  for (int i = 0; i < 3; ++i)
    out[(size_t)row * 768 + tid * 3 + i] = (__bf16)(vals[i] * scale * w[tid * 3 + i]);
}

// RMS + positional add: t = rms(in)*w + pos  (f32 out). One block per row.
__global__ __launch_bounds__(256) void rms_posadd_kernel(
    const float* __restrict__ in, const float* __restrict__ w,
    const float* __restrict__ pos, float* __restrict__ out) {
  __shared__ float red[256];
  int row = blockIdx.x, tid = threadIdx.x;
  int l = row & 255;
  const float* rp = in + (size_t)row * 768;
  float vals[3];
  float ss = 0.f;
#pragma unroll
  for (int i = 0; i < 3; ++i) {
    float v = rp[tid * 3 + i];
    vals[i] = v;
    ss += v * v;
  }
  red[tid] = ss;
  __syncthreads();
  for (int s = 128; s > 0; s >>= 1) {
    if (tid < s) red[tid] += red[tid + s];
    __syncthreads();
  }
  float scale = rsqrtf(red[0] / 768.0f + 1e-5f);
#pragma unroll
  for (int i = 0; i < 3; ++i) {
    int col = tid * 3 + i;
    out[(size_t)row * 768 + col] =
        vals[i] * scale * w[col] + pos[(size_t)l * 768 + col];
  }
}

// GLU + mish: H[1024,6144] -> G_bf16[1024,3072]
__global__ void glu_kernel(const float* __restrict__ H, __bf16* __restrict__ G, int n) {
  int i = blockIdx.x * 256 + threadIdx.x;
  if (i >= n) return;
  int row = i / FF, j = i % FF;
  float a = H[(size_t)row * 2 * FF + j];
  float g = H[(size_t)row * 2 * FF + FF + j];
  G[i] = (__bf16)(mish_f(a) * g);
}

__global__ void copy_f32_kernel(const float* __restrict__ a, float* __restrict__ o, int n) {
  int i = blockIdx.x * 256 + threadIdx.x;
  if (i < n) o[i] = a[i];
}

__global__ void add_f32_kernel(const float* __restrict__ a, float* __restrict__ o, int n) {
  int i = blockIdx.x * 256 + threadIdx.x;
  if (i < n) o[i] += a[i];
}

// cat[row, 0:768)=bf16(t), [768:1536)=bf16(skip)
__global__ void cat_pack_kernel(const float* __restrict__ t,
                                const float* __restrict__ skip,
                                __bf16* __restrict__ cat, int n) {
  int i = blockIdx.x * 256 + threadIdx.x;
  if (i >= n) return;
  int row = i / 1536, col = i % 1536;
  float v = (col < 768) ? t[(size_t)row * 768 + col]
                        : skip[(size_t)row * 768 + col - 768];
  cat[i] = (__bf16)v;
}

// proj[1024,768] -> img[B,3,256,256]
__global__ void unpatchify_kernel(const float* __restrict__ proj,
                                  float* __restrict__ img, int n) {
  int i = blockIdx.x * 256 + threadIdx.x;
  if (i >= n) return;
  int x = i & 255, y = (i >> 8) & 255;
  int c = (i >> 16) % 3, b = i / (3 * 65536);
  int gy = y >> 4, p1 = y & 15, gx = x >> 4, p2 = x & 15;
  img[i] = proj[(size_t)(b * 256 + gy * 16 + gx) * 768 + (p1 * 16 + p2) * 3 + c];
}

DEV int refl(int i, int n) {
  if (i < 0) i = -i;
  if (i >= n) i = 2 * n - 2 - i;
  return i;
}

// conv 3->128, 3x3, reflect pad 1
__global__ void conv_in_kernel(const float* __restrict__ img,
                               const float* __restrict__ wgt,
                               float* __restrict__ out, int n) {
  int i = blockIdx.x * 256 + threadIdx.x;
  if (i >= n) return;
  int x = i & 255, y = (i >> 8) & 255, oc = (i >> 16) & 127, b = i >> 23;
  float s = 0.f;
#pragma unroll
  for (int ic = 0; ic < 3; ++ic)
#pragma unroll
    for (int ky = 0; ky < 3; ++ky)
#pragma unroll
      for (int kx = 0; kx < 3; ++kx) {
        int yy = refl(y + ky - 1, 256), xx = refl(x + kx - 1, 256);
        s += img[((size_t)(b * 3 + ic) * 256 + yy) * 256 + xx] *
             wgt[((oc * 3 + ic) * 3 + ky) * 3 + kx];
      }
  out[i] = s;
}

// depthwise 5x5, reflect pad 2*dil, dilation dil
__global__ void dwconv_kernel(const float* __restrict__ in,
                              const float* __restrict__ wgt,
                              float* __restrict__ out, int dil, int n) {
  int i = blockIdx.x * 256 + threadIdx.x;
  if (i >= n) return;
  int x = i & 255, y = (i >> 8) & 255, c = (i >> 16) & 127, b = i >> 23;
  const float* base = in + ((size_t)(b * 128 + c) << 16);
  float s = 0.f;
#pragma unroll
  for (int ky = 0; ky < 5; ++ky)
#pragma unroll
    for (int kx = 0; kx < 5; ++kx) {
      int yy = refl(y + (ky - 2) * dil, 256);
      int xx = refl(x + (kx - 2) * dil, 256);
      s += base[yy * 256 + xx] * wgt[c * 25 + ky * 5 + kx];
    }
  out[i] = s;
}

// GroupNorm stats: one block per (b,g) of 32; group = 16 ch * 65536 px
__global__ __launch_bounds__(256) void gn_stats_kernel(const float* __restrict__ in,
                                                       float* __restrict__ stats) {
  __shared__ float r1[256], r2[256];
  int bg = blockIdx.x;
  int b = bg >> 3, g = bg & 7;
  int tid = threadIdx.x;
  float s = 0.f, s2 = 0.f;
  for (int idx = tid; idx < 16 * 65536; idx += 256) {
    int c = g * 16 + (idx >> 16);
    int pix = idx & 65535;
    float v = in[((size_t)(b * 128 + c) << 16) + pix];
    s += v;
    s2 += v * v;
  }
  r1[tid] = s;
  r2[tid] = s2;
  __syncthreads();
  for (int st = 128; st > 0; st >>= 1) {
    if (tid < st) { r1[tid] += r1[tid + st]; r2[tid] += r2[tid + st]; }
    __syncthreads();
  }
  if (tid == 0) {
    float inv_n = 1.0f / (16.0f * 65536.0f);
    float mean = r1[0] * inv_n;
    float var = r2[0] * inv_n - mean * mean;
    stats[2 * bg] = mean;
    stats[2 * bg + 1] = rsqrtf(var + 1e-5f);
  }
}

__global__ void gn_apply_mish_kernel(const float* __restrict__ in,
                                     float* __restrict__ out,
                                     const float* __restrict__ stats,
                                     const float* __restrict__ w,
                                     const float* __restrict__ bias, int n) {
  int i = blockIdx.x * 256 + threadIdx.x;
  if (i >= n) return;
  int c = (i >> 16) & 127, b = i >> 23;
  int bg = b * 8 + (c >> 4);
  float mean = stats[2 * bg], rstd = stats[2 * bg + 1];
  float v = (in[i] - mean) * rstd * w[c] + bias[c];
  out[i] = mish_f(v);
}

// head conv 128->1, 3x3, reflect pad 1
__global__ void head_conv_kernel(const float* __restrict__ in,
                                 const float* __restrict__ wgt,
                                 float* __restrict__ out, int n) {
  int i = blockIdx.x * 256 + threadIdx.x;
  if (i >= n) return;
  int x = i & 255, y = (i >> 8) & 255, b = i >> 16;
  float s = 0.f;
  for (int ic = 0; ic < 128; ++ic) {
    const float* base = in + ((size_t)(b * 128 + ic) << 16);
#pragma unroll
    for (int ky = 0; ky < 3; ++ky)
#pragma unroll
      for (int kx = 0; kx < 3; ++kx) {
        int yy = refl(y + ky - 1, 256), xx = refl(x + kx - 1, 256);
        s += base[yy * 256 + xx] * wgt[(ic * 3 + ky) * 3 + kx];
      }
  }
  out[i] = s;
}

// ---------------------------------------------------------------------------
// Host orchestration
// ---------------------------------------------------------------------------
#define CEIL(a, b) (((a) + (b) - 1) / (b))

static const size_t ATTN_LDS =
    (3 * 256 * 64 + 8 * 16 * 256) * 2;  // Q,K,Vt + per-wave P => 163840 B

struct Work {
  float* t; float* skip; __bf16* xn; float* qkv; float* H; __bf16* G;
  __bf16* aO; float* proj; float* img; float* stats; __bf16* Wt;
  float* P; float* Q;
};

static void gemm(hipStream_t s, Work& ws, const __bf16* A, const float* W,
                 const float* bias, const float* resid, float* out,
                 int M, int N, int K) {
  dim3 gp(N / 32, K / 32);
  transpose_pack_kernel<<<gp, 256, 0, s>>>(W, ws.Wt, K, N);
  dim3 grid(N / 128, M / 64);
  gemm_kernel<<<grid, 256, 0, s>>>(A, ws.Wt, bias, resid, out, M, N, K);
}

static void run_block(hipStream_t s, Work& ws, const float* anw, const float* qkvw,
                      const float* ow, const float* fnw, const float* w1,
                      const float* w2) {
  rms_kernel<<<MT, 256, 0, s>>>(ws.t, anw, ws.xn);
  gemm(s, ws, ws.xn, qkvw, nullptr, nullptr, ws.qkv, MT, 3 * Dm, Dm);
  attn_kernel<<<Bv * NH, 256, ATTN_LDS, s>>>(ws.qkv, ws.aO);
  gemm(s, ws, ws.aO, ow, nullptr, ws.t, ws.t, MT, Dm, Dm);
  rms_kernel<<<MT, 256, 0, s>>>(ws.t, fnw, ws.xn);
  gemm(s, ws, ws.xn, w1, nullptr, nullptr, ws.H, MT, 2 * FF, Dm);
  glu_kernel<<<CEIL(MT * FF, 256), 256, 0, s>>>(ws.H, ws.G, MT * FF);
  gemm(s, ws, ws.G, w2, nullptr, ws.t, ws.t, MT, Dm, FF);
}

extern "C" void kernel_launch(void* const* d_in, const int* in_sizes, int n_in,
                              void* d_out, int out_size, void* d_ws, size_t ws_size,
                              hipStream_t stream) {
  (void)in_sizes; (void)n_in; (void)out_size; (void)ws_size;
  const float* x            = (const float*)d_in[0];
  const float* pe_norm_w    = (const float*)d_in[1];
  const float* pe_proj_w    = (const float*)d_in[2];
  const float* pe_proj_b    = (const float*)d_in[3];
  const float* pe_norm_out_w= (const float*)d_in[4];
  const float* pos_embed    = (const float*)d_in[5];
  const float* in_attn_norm = (const float*)d_in[6];
  const float* in_qkv_w     = (const float*)d_in[7];
  const float* in_out_w     = (const float*)d_in[8];
  const float* in_ff_norm   = (const float*)d_in[9];
  const float* in_w1        = (const float*)d_in[10];
  const float* in_w2        = (const float*)d_in[11];
  const float* mid_attn_norm= (const float*)d_in[12];
  const float* mid_qkv_w    = (const float*)d_in[13];
  const float* mid_out_w    = (const float*)d_in[14];
  const float* mid_ff_norm  = (const float*)d_in[15];
  const float* mid_w1       = (const float*)d_in[16];
  const float* mid_w2       = (const float*)d_in[17];
  const float* out_skip_w   = (const float*)d_in[18];
  const float* out_skip_b   = (const float*)d_in[19];
  const float* out_attn_norm= (const float*)d_in[20];
  const float* out_qkv_w    = (const float*)d_in[21];
  const float* out_out_w    = (const float*)d_in[22];
  const float* out_ff_norm  = (const float*)d_in[23];
  const float* out_w1       = (const float*)d_in[24];
  const float* out_w2       = (const float*)d_in[25];
  const float* final_norm_w = (const float*)d_in[26];
  const float* ol_proj_w    = (const float*)d_in[27];
  const float* ol_proj_b    = (const float*)d_in[28];
  const float* rb_conv_in_w = (const float*)d_in[29];
  const float* rb_gn_w      = (const float*)d_in[30];
  const float* rb_gn_b      = (const float*)d_in[31];
  const float* rb_conv1_w   = (const float*)d_in[32];
  const float* rb_conv2_w   = (const float*)d_in[33];
  const float* head_conv_w  = (const float*)d_in[34];

  // Workspace carve-up (deterministic, 256B aligned)
  char* base = (char*)d_ws;
  size_t off = 0;
  auto alloc = [&](size_t bytes) -> char* {
    char* p = base + off;
    off = (off + bytes + 255) & ~(size_t)255;
    return p;
  };
  Work ws;
  ws.t     = (float*)alloc((size_t)MT * Dm * 4);
  ws.skip  = (float*)alloc((size_t)MT * Dm * 4);
  ws.xn    = (__bf16*)alloc((size_t)MT * 1536 * 2);
  ws.qkv   = (float*)alloc((size_t)MT * 3 * Dm * 4);
  ws.H     = (float*)alloc((size_t)MT * 2 * FF * 4);
  ws.G     = (__bf16*)alloc((size_t)MT * FF * 2);
  ws.aO    = (__bf16*)alloc((size_t)MT * Dm * 2);
  ws.proj  = (float*)alloc((size_t)MT * Dm * 4);
  ws.img   = (float*)alloc((size_t)Bv * 3 * 256 * 256 * 4);
  ws.stats = (float*)alloc(64 * 4);
  ws.Wt    = (__bf16*)alloc((size_t)Dm * 2 * FF * 2);  // max K*N bf16
  ws.P     = (float*)alloc((size_t)NCONV * 4);
  ws.Q     = (float*)alloc((size_t)NCONV * 4);

  // ---- Patch embed ----
  patchify_rms_kernel<<<MT, 256, 0, stream>>>(x, pe_norm_w, ws.xn);
  gemm(stream, ws, ws.xn, pe_proj_w, pe_proj_b, nullptr, ws.H, MT, Dm, Dm);
  rms_posadd_kernel<<<MT, 256, 0, stream>>>(ws.H, pe_norm_out_w, pos_embed, ws.t);

  // ---- 6 in-blocks ----
  for (int i = 0; i < 6; ++i)
    run_block(stream, ws, in_attn_norm + (size_t)i * Dm,
              in_qkv_w + (size_t)i * Dm * 3 * Dm, in_out_w + (size_t)i * Dm * Dm,
              in_ff_norm + (size_t)i * Dm, in_w1 + (size_t)i * Dm * 2 * FF,
              in_w2 + (size_t)i * FF * Dm);
  copy_f32_kernel<<<CEIL(MT * Dm, 256), 256, 0, stream>>>(ws.t, ws.skip, MT * Dm);

  // ---- 3 mid-blocks ----
  for (int i = 0; i < 3; ++i)
    run_block(stream, ws, mid_attn_norm + (size_t)i * Dm,
              mid_qkv_w + (size_t)i * Dm * 3 * Dm, mid_out_w + (size_t)i * Dm * Dm,
              mid_ff_norm + (size_t)i * Dm, mid_w1 + (size_t)i * Dm * 2 * FF,
              mid_w2 + (size_t)i * FF * Dm);

  // ---- skip fusion + out-block ----
  cat_pack_kernel<<<CEIL(MT * 1536, 256), 256, 0, stream>>>(ws.t, ws.skip, ws.xn,
                                                            MT * 1536);
  gemm(stream, ws, ws.xn, out_skip_w, out_skip_b, nullptr, ws.t, MT, Dm, 2 * Dm);
  run_block(stream, ws, out_attn_norm, out_qkv_w, out_out_w, out_ff_norm, out_w1,
            out_w2);

  // ---- final norm + out proj + unpatchify ----
  rms_kernel<<<MT, 256, 0, stream>>>(ws.t, final_norm_w, ws.xn);
  gemm(stream, ws, ws.xn, ol_proj_w, ol_proj_b, nullptr, ws.proj, MT, Dm, Dm);
  unpatchify_kernel<<<CEIL(Bv * 3 * 65536, 256), 256, 0, stream>>>(
      ws.proj, ws.img, Bv * 3 * 65536);

  // ---- ResNet tail ----
  int nb = CEIL(NCONV, 256);
  conv_in_kernel<<<nb, 256, 0, stream>>>(ws.img, rb_conv_in_w, ws.P, NCONV);
  gn_stats_kernel<<<32, 256, 0, stream>>>(ws.P, ws.stats);
  gn_apply_mish_kernel<<<nb, 256, 0, stream>>>(ws.P, ws.P, ws.stats, rb_gn_w + 0,
                                               rb_gn_b + 0, NCONV);
  dwconv_kernel<<<nb, 256, 0, stream>>>(ws.P, rb_conv1_w, ws.Q, 1, NCONV);
  gn_stats_kernel<<<32, 256, 0, stream>>>(ws.Q, ws.stats);
  gn_apply_mish_kernel<<<nb, 256, 0, stream>>>(ws.Q, ws.Q, ws.stats, rb_gn_w + 128,
                                               rb_gn_b + 128, NCONV);
  add_f32_kernel<<<nb, 256, 0, stream>>>(ws.P, ws.Q, NCONV);
  gn_stats_kernel<<<32, 256, 0, stream>>>(ws.Q, ws.stats);
  gn_apply_mish_kernel<<<nb, 256, 0, stream>>>(ws.Q, ws.P, ws.stats, rb_gn_w + 256,
                                               rb_gn_b + 256, NCONV);
  dwconv_kernel<<<nb, 256, 0, stream>>>(ws.P, rb_conv2_w, ws.Q, 3, NCONV);
  gn_stats_kernel<<<32, 256, 0, stream>>>(ws.Q, ws.stats);
  gn_apply_mish_kernel<<<nb, 256, 0, stream>>>(ws.Q, ws.Q, ws.stats, rb_gn_w + 384,
                                               rb_gn_b + 384, NCONV);
  add_f32_kernel<<<nb, 256, 0, stream>>>(ws.P, ws.Q, NCONV);
  gn_stats_kernel<<<32, 256, 0, stream>>>(ws.Q, ws.stats);
  gn_apply_mish_kernel<<<nb, 256, 0, stream>>>(ws.Q, ws.Q, ws.stats, rb_gn_w + 512,
                                               rb_gn_b + 512, NCONV);
  head_conv_kernel<<<CEIL(Bv * 65536, 256), 256, 0, stream>>>(
      ws.Q, head_conv_w, (float*)d_out, Bv * 65536);
}
